// GIMO_ADT_Autoregressive_Model_16492674417341
// MI455X (gfx1250) — compile-verified, matrix operand order checked by reading, not run on hardware
//
#include <hip/hip_runtime.h>
#include <hip/hip_bf16.h>

// ---------------------------------------------------------------------------
// Transformer block forward (B=2, S=2048, D=1024, H=16, HD=64, FF=4096)
// bf16 WMMA (v_wmma_f32_16x16x32_bf16) for all GEMMs + flash attention.
// GEMM tiles staged into LDS by the Tensor Data Mover (tensor_load_to_lds,
// TENSORcnt-tracked) with D# LDS padding reproducing the 48-element stride.
// ---------------------------------------------------------------------------

typedef __attribute__((ext_vector_type(16))) __bf16       v16bf;
typedef __attribute__((ext_vector_type(8)))  float        v8f;
typedef __attribute__((ext_vector_type(4)))  unsigned int u32x4;
typedef __attribute__((ext_vector_type(4)))  unsigned int v4u;
typedef __attribute__((ext_vector_type(8)))  int          v8i;
typedef __attribute__((ext_vector_type(4)))  int          v4i;

#define B_   2
#define S_   2048
#define D_   1024
#define H_   16
#define HD_  64
#define FF_  4096
#define MT_  (B_ * S_)          // 4096 rows of the activation matrix

static __device__ __forceinline__ unsigned short f2bf(float f) {
    unsigned int u = __float_as_uint(f);
    u += 0x7FFFu + ((u >> 16) & 1u);          // round-to-nearest-even
    return (unsigned short)(u >> 16);
}
static __device__ __forceinline__ float bf2f(unsigned short h) {
    return __uint_as_float(((unsigned int)h) << 16);
}

// ---------------- TDM: 2D bf16 tile load, global -> padded LDS -------------
// Tile: rows x 32 bf16 elements (64B/row), LDS pitch 96B (pad 32B after each
// 64B row => 48-element stride). D# layout per CDNA5 ISA 08_async_tensor §8.
// This toolchain exposes the 6-arg builtin:
//   (uint32x4 g0, int32x8 g1, int32x4 g2, int32x4 g3, int32x8 extra, i32 cpol)
#if __has_builtin(__builtin_amdgcn_tensor_load_to_lds)
#define USE_TDM 1
static __device__ __forceinline__ void tdm_load_tile(const void* gptr,
                                                     unsigned lds_addr,
                                                     int rows, int strideElems) {
    unsigned long long ga = (unsigned long long)(uintptr_t)gptr;
    // group0: count=1 | lds_addr | global_addr[56:0] | type=2
    v4u g0 = { 1u,
               lds_addr,
               (unsigned)ga,
               ((unsigned)((ga >> 32) & 0x01FFFFFFull)) | 0x80000000u };
    // group1: data_size=1(2B), pad_enable, pad_interval=3(64B), pad_amount=7(32B)
    //         tensor_dim0/1 = huge (no clip), tile_dim0=32, tile_dim1=rows,
    //         tensor_dim0_stride = strideElems
    v8i g1 = { (int)0x0ED10000u,                 // flags (see above)
               (int)0xFFFF0000u,                 // tensor_dim0[15:0] << 16
               (int)0xFFFF7FFFu,                 // td0[31:16] | td1[15:0]<<16
               (int)(0x7FFFu | (32u << 16)),     // td1[31:16] | tile_dim0<<16
               rows,                             // tile_dim1 (tile_dim2 = 0)
               strideElems,                      // tensor_dim0_stride[31:0]
               0, 0 };
    v4i z4 = { 0, 0, 0, 0 };
    v8i z8 = { 0, 0, 0, 0, 0, 0, 0, 0 };
    __builtin_amdgcn_tensor_load_to_lds(g0, g1, z4, z4, z8, 0);
}
#else
#define USE_TDM 0
#endif

// -------------------------- fp32 -> bf16 cast ------------------------------
__global__ void cvt_bf16_kernel(const float* __restrict__ src,
                                unsigned short* __restrict__ dst, int n) {
    int base = (blockIdx.x * blockDim.x + threadIdx.x) * 4;
    if (base >= n) return;
    unsigned int a = ((unsigned int)f2bf(src[base + 1]) << 16) | f2bf(src[base + 0]);
    unsigned int b = ((unsigned int)f2bf(src[base + 3]) << 16) | f2bf(src[base + 2]);
    *(unsigned int*)(dst + base)     = a;
    *(unsigned int*)(dst + base + 2) = b;
}

// ------------------------------ LayerNorm ----------------------------------
__global__ __launch_bounds__(256) void ln_kernel(const float* __restrict__ x,
                                                 const float* __restrict__ g,
                                                 const float* __restrict__ b,
                                                 unsigned short* __restrict__ out) {
    int row = blockIdx.x, tid = threadIdx.x;
    const float* xr = x + (size_t)row * D_;
    float v[4]; float s = 0.f;
#pragma unroll
    for (int i = 0; i < 4; ++i) { v[i] = xr[tid + i * 256]; s += v[i]; }
    __shared__ float sb[8], sb2[8];
    int wid = tid >> 5, lane = tid & 31;
#pragma unroll
    for (int m = 16; m >= 1; m >>= 1) s += __shfl_xor(s, m, 32);
    if (lane == 0) sb[wid] = s;
    __syncthreads();
    float tot = 0.f;
#pragma unroll
    for (int w = 0; w < 8; ++w) tot += sb[w];
    float mean = tot * (1.f / D_);
    float vs = 0.f;
#pragma unroll
    for (int i = 0; i < 4; ++i) { float d = v[i] - mean; vs += d * d; }
#pragma unroll
    for (int m = 16; m >= 1; m >>= 1) vs += __shfl_xor(vs, m, 32);
    if (lane == 0) sb2[wid] = vs;
    __syncthreads();
    float vt = 0.f;
#pragma unroll
    for (int w = 0; w < 8; ++w) vt += sb2[w];
    float inv = rsqrtf(vt * (1.f / D_) + 1e-5f);
#pragma unroll
    for (int i = 0; i < 4; ++i) {
        int c = tid + i * 256;
        out[(size_t)row * D_ + c] = f2bf((v[i] - mean) * inv * g[c] + b[c]);
    }
}

// -------------------------------- RoPE -------------------------------------
__global__ void rope_kernel(unsigned short* __restrict__ q,
                            const float* __restrict__ freqs) {
    int idx = blockIdx.x * blockDim.x + threadIdx.x;   // B*S*H*32 threads
    int i = idx & 31;
    int r = idx >> 5;                                  // (b*S+s)*H + h
    int s = (r / H_) % S_;
    float f = freqs[s * 32 + i];
    float c = __cosf(f), sn = __sinf(f);
    size_t base = (size_t)r * 64 + 2 * i;
    float e = bf2f(q[base]), o = bf2f(q[base + 1]);
    q[base]     = f2bf(e * c - o * sn);
    q[base + 1] = f2bf(o * c + e * sn);
}

// ------------------------- WMMA GEMM: out = A * W^T ------------------------
// A: MxK bf16 row-major, W: NxK bf16 row-major. 256 threads = 8 waves;
// block tile 128(M) x 64(N), BK=32; each wave computes 2x2 WMMA tiles.
// Tiles staged into LDS by the TDM (wave 0 issues, s_wait_tensorcnt, barrier).
template <int ACT, bool HAS_RES, bool OUT_F, bool OUT_B>
__global__ __launch_bounds__(256) void gemm_wmma_kernel(
    const unsigned short* __restrict__ A,
    const unsigned short* __restrict__ W,
    const float* __restrict__ bias,
    const float* __restrict__ res,
    float* __restrict__ outF,
    unsigned short* __restrict__ outB,
    int M, int N, int K) {
    (void)M;
    const int LDA = 48;                        // LDS row stride (elems), 96B
    __shared__ unsigned short As[128 * 48];
    __shared__ unsigned short Bs[64 * 48];
    int tid  = threadIdx.x;
    int wave = tid >> 5, lane = tid & 31;
    int m0 = (wave >> 1) * 32;
    int n0 = (wave & 1) * 32;
    int blockM = blockIdx.y * 128;
    int blockN = blockIdx.x * 64;

    v8f zero = {};
    v8f acc[2][2];
#pragma unroll
    for (int mi = 0; mi < 2; ++mi)
#pragma unroll
        for (int ni = 0; ni < 2; ++ni) acc[mi][ni] = zero;

    int fr = lane & 15;
    int kh = (lane >> 4) * 8;
#if USE_TDM
    unsigned asAddr = (unsigned)(uintptr_t)(void*)As;
    unsigned bsAddr = (unsigned)(uintptr_t)(void*)Bs;
    const unsigned short* aBase = A + (size_t)blockM * K;
    const unsigned short* wBase = W + (size_t)blockN * K;
#else
    int rL = tid >> 2;                         // 0..63
    int cL = (tid & 3) * 8;                    // 0,8,16,24
#endif

    for (int k0 = 0; k0 < K; k0 += 32) {
#if USE_TDM
        if (tid < 32) {                        // wave 0 drives the TDM
            tdm_load_tile(aBase + k0, asAddr, 128, K);
            tdm_load_tile(wBase + k0, bsAddr, 64, K);
            __builtin_amdgcn_s_wait_tensorcnt(0);
        }
        __syncthreads();
#else
#pragma unroll
        for (int i = 0; i < 2; ++i) {          // A tile: 128x32
            int r = rL + i * 64;
            const unsigned short* gp = A + (size_t)(blockM + r) * K + k0 + cL;
            *(u32x4*)&As[r * LDA + cL] = *(const u32x4*)gp;
        }
        {                                      // B tile: 64x32 (rows = N)
            const unsigned short* gp = W + (size_t)(blockN + rL) * K + k0 + cL;
            *(u32x4*)&Bs[rL * LDA + cL] = *(const u32x4*)gp;
        }
        __syncthreads();
#endif

        v16bf af[2], bfv[2];
#pragma unroll
        for (int mi = 0; mi < 2; ++mi) {
            int r = m0 + mi * 16 + fr;
            union { v16bf v; u32x4 q[2]; } u;
            u.q[0] = *(const u32x4*)&As[r * LDA + kh];
            u.q[1] = *(const u32x4*)&As[r * LDA + kh + 16];
            af[mi] = u.v;
        }
#pragma unroll
        for (int ni = 0; ni < 2; ++ni) {
            int r = n0 + ni * 16 + fr;
            union { v16bf v; u32x4 q[2]; } u;
            u.q[0] = *(const u32x4*)&Bs[r * LDA + kh];
            u.q[1] = *(const u32x4*)&Bs[r * LDA + kh + 16];
            bfv[ni] = u.v;
        }
#pragma unroll
        for (int mi = 0; mi < 2; ++mi)
#pragma unroll
            for (int ni = 0; ni < 2; ++ni)
                acc[mi][ni] = __builtin_amdgcn_wmma_f32_16x16x32_bf16(
                    false, af[mi], false, bfv[ni], (short)0, acc[mi][ni],
                    false, false);
        __syncthreads();
    }

    // epilogue: C-layout = lane holds col (lane&15), rows i + 8*(lane>>4)
    int nl = lane & 15, laneHi = lane >> 4;
#pragma unroll
    for (int mi = 0; mi < 2; ++mi) {
#pragma unroll
        for (int ni = 0; ni < 2; ++ni) {
            int gn = blockN + n0 + ni * 16 + nl;
            float bv = bias[gn];
#pragma unroll
            for (int i = 0; i < 8; ++i) {
                int gm = blockM + m0 + mi * 16 + i + 8 * laneHi;
                float val = acc[mi][ni][i] + bv;
                if (ACT == 1)  // exact GELU
                    val = 0.5f * val * (1.f + erff(val * 0.70710678118654752f));
                size_t idx = (size_t)gm * N + gn;
                if (HAS_RES) val += res[idx];
                if (OUT_F) outF[idx] = val;
                if (OUT_B) outB[idx] = f2bf(val);
            }
        }
    }
}

// ----------------------- Flash attention (per head) ------------------------
// grid: (S/16, B*H), one wave/block. q/k/v bf16 viewed as (B*S*H, 64).
__global__ __launch_bounds__(32) void attn_kernel(
    const unsigned short* __restrict__ qb,
    const unsigned short* __restrict__ kb,
    const unsigned short* __restrict__ vb,
    const unsigned char* __restrict__ pad,
    unsigned short* __restrict__ ob) {
    __shared__ unsigned short pbuf[16 * 32];
    __shared__ unsigned short vT[64 * 32];
    int lane = threadIdx.x;
    int q0 = blockIdx.x * 16;
    int h = blockIdx.y % H_, b = blockIdx.y / H_;
    const int RS = H_ * HD_;                              // 1024 elems / seq step
    size_t base = ((size_t)b * S_) * RS + (size_t)h * HD_;
    int fr = lane & 15, kh = (lane >> 4) * 8, laneHi = lane >> 4;

    v16bf qf[2];
#pragma unroll
    for (int dc = 0; dc < 2; ++dc) {
        const unsigned short* p = qb + base + (size_t)(q0 + fr) * RS + dc * 32 + kh;
        union { v16bf v; u32x4 q[2]; } u;
        u.q[0] = *(const u32x4*)p;
        u.q[1] = *(const u32x4*)(p + 16);
        qf[dc] = u.v;
    }

    v8f zero = {};
    float rm[8], rs[8];
    v8f o[4];
#pragma unroll
    for (int i = 0; i < 8; ++i) { rm[i] = -1e30f; rs[i] = 0.f; }
#pragma unroll
    for (int t = 0; t < 4; ++t) o[t] = zero;
    int nl = fr;

    for (int j0 = 0; j0 <= q0 + 15; j0 += 32) {
        // ---- scores: two 16x16 tiles over d=64 (2 WMMAs each) ----
        v8f sc[2];
#pragma unroll
        for (int t = 0; t < 2; ++t) {
            sc[t] = zero;
#pragma unroll
            for (int dc = 0; dc < 2; ++dc) {
                const unsigned short* p =
                    kb + base + (size_t)(j0 + t * 16 + fr) * RS + dc * 32 + kh;
                union { v16bf v; u32x4 q[2]; } u;
                u.q[0] = *(const u32x4*)p;
                u.q[1] = *(const u32x4*)(p + 16);
                sc[t] = __builtin_amdgcn_wmma_f32_16x16x32_bf16(
                    false, qf[dc], false, u.v, (short)0, sc[t], false, false);
            }
        }
        // ---- stage V chunk transposed: vT[d][k] ----
        {
            const unsigned short* vp = vb + base + (size_t)(j0 + lane) * RS;
#pragma unroll
            for (int d = 0; d < 64; ++d) vT[d * 32 + lane] = vp[d];
        }
        __syncthreads();
        // ---- mask + online softmax (fp32) ----
        float p0[8], p1[8];
#pragma unroll
        for (int i = 0; i < 8; ++i) {
            int sq = q0 + i + 8 * laneHi;
            float s0 = sc[0][i] * 0.125f;                 // 1/sqrt(64)
            float s1 = sc[1][i] * 0.125f;
            int k0i = j0 + nl, k1i = j0 + 16 + nl;
            if (k0i > sq || pad[b * S_ + k0i]) s0 = -1e9f;
            if (k1i > sq || pad[b * S_ + k1i]) s1 = -1e9f;
            float cm = fmaxf(s0, s1);
#pragma unroll
            for (int m = 8; m >= 1; m >>= 1) cm = fmaxf(cm, __shfl_xor(cm, m, 32));
            float nm = fmaxf(rm[i], cm);
            float alpha = __expf(rm[i] - nm);
            float e0 = __expf(s0 - nm), e1 = __expf(s1 - nm);
            float psum = e0 + e1;
#pragma unroll
            for (int m = 8; m >= 1; m >>= 1) psum += __shfl_xor(psum, m, 32);
            rs[i] = rs[i] * alpha + psum;
            rm[i] = nm;
#pragma unroll
            for (int t = 0; t < 4; ++t) o[t][i] *= alpha;
            p0[i] = e0; p1[i] = e1;
        }
        // ---- re-layout P (C-layout -> A-fragment) through LDS ----
#pragma unroll
        for (int i = 0; i < 8; ++i) {
            int m = i + 8 * laneHi;
            pbuf[m * 32 + nl]      = f2bf(p0[i]);
            pbuf[m * 32 + 16 + nl] = f2bf(p1[i]);
        }
        __syncthreads();
        v16bf pf;
        {
            union { v16bf v; u32x4 q[2]; } u;
            u.q[0] = *(const u32x4*)&pbuf[fr * 32 + kh];
            u.q[1] = *(const u32x4*)&pbuf[fr * 32 + kh + 16];
            pf = u.v;
        }
        // ---- P·V: 4 d-tiles of 16, K=32 keys ----
#pragma unroll
        for (int dt = 0; dt < 4; ++dt) {
            union { v16bf v; u32x4 q[2]; } u;
            int d = dt * 16 + fr;
            u.q[0] = *(const u32x4*)&vT[d * 32 + kh];
            u.q[1] = *(const u32x4*)&vT[d * 32 + kh + 16];
            o[dt] = __builtin_amdgcn_wmma_f32_16x16x32_bf16(
                false, pf, false, u.v, (short)0, o[dt], false, false);
        }
        __syncthreads();
    }
    // ---- normalize + store bf16 ----
#pragma unroll
    for (int i = 0; i < 8; ++i) {
        int sq = q0 + i + 8 * laneHi;
        float inv = 1.f / rs[i];
        size_t rb = base + (size_t)sq * RS;
#pragma unroll
        for (int dt = 0; dt < 4; ++dt)
            ob[rb + dt * 16 + nl] = f2bf(o[dt][i] * inv);
    }
}

// ------------------------------ orchestration ------------------------------
extern "C" void kernel_launch(void* const* d_in, const int* in_sizes, int n_in,
                              void* d_out, int out_size, void* d_ws, size_t ws_size,
                              hipStream_t stream) {
    (void)in_sizes; (void)n_in; (void)out_size; (void)ws_size;
    const float* x      = (const float*)d_in[0];
    const float* freqs  = (const float*)d_in[1];
    const unsigned char* pad = (const unsigned char*)d_in[2];
    const float* ln1_g  = (const float*)d_in[3];
    const float* ln1_b  = (const float*)d_in[4];
    const float* Wq     = (const float*)d_in[5];
    const float* bq     = (const float*)d_in[6];
    const float* Wk     = (const float*)d_in[7];
    const float* bk     = (const float*)d_in[8];
    const float* Wv     = (const float*)d_in[9];
    const float* bv     = (const float*)d_in[10];
    const float* Wo     = (const float*)d_in[11];
    const float* bo     = (const float*)d_in[12];
    const float* ln2_g  = (const float*)d_in[13];
    const float* ln2_b  = (const float*)d_in[14];
    const float* W1     = (const float*)d_in[15];
    const float* b1     = (const float*)d_in[16];
    const float* W2     = (const float*)d_in[17];
    const float* b2     = (const float*)d_in[18];

    char* ws = (char*)d_ws;
    size_t off = 0;
    auto take = [&](size_t bytes) {
        void* p = ws + off;
        off += (bytes + 255) & ~(size_t)255;
        return p;
    };
    unsigned short* wq_b = (unsigned short*)take((size_t)D_ * D_ * 2);
    unsigned short* wk_b = (unsigned short*)take((size_t)D_ * D_ * 2);
    unsigned short* wv_b = (unsigned short*)take((size_t)D_ * D_ * 2);
    unsigned short* wo_b = (unsigned short*)take((size_t)D_ * D_ * 2);
    unsigned short* w1_b = (unsigned short*)take((size_t)FF_ * D_ * 2);
    unsigned short* w2_b = (unsigned short*)take((size_t)FF_ * D_ * 2);
    unsigned short* xn_b = (unsigned short*)take((size_t)MT_ * D_ * 2);
    unsigned short* q_b  = (unsigned short*)take((size_t)MT_ * D_ * 2);
    unsigned short* k_b  = (unsigned short*)take((size_t)MT_ * D_ * 2);
    unsigned short* v_b  = (unsigned short*)take((size_t)MT_ * D_ * 2);
    unsigned short* at_b = (unsigned short*)take((size_t)MT_ * D_ * 2);
    float*          x2_f = (float*)take((size_t)MT_ * D_ * 4);
    unsigned short* h_b  = (unsigned short*)take((size_t)MT_ * D_ * 2);
    unsigned short* g_b  = (unsigned short*)take((size_t)MT_ * FF_ * 2);

    auto cvt = [&](const float* s, unsigned short* d, int n) {
        cvt_bf16_kernel<<<(n / 4 + 255) / 256, 256, 0, stream>>>(s, d, n);
    };
    cvt(Wq, wq_b, D_ * D_);
    cvt(Wk, wk_b, D_ * D_);
    cvt(Wv, wv_b, D_ * D_);
    cvt(Wo, wo_b, D_ * D_);
    cvt(W1, w1_b, FF_ * D_);
    cvt(W2, w2_b, FF_ * D_);

    // LN1
    ln_kernel<<<MT_, 256, 0, stream>>>(x, ln1_g, ln1_b, xn_b);

    // QKV projections
    dim3 gD(D_ / 64, MT_ / 128);
    gemm_wmma_kernel<0, false, false, true><<<gD, 256, 0, stream>>>(
        xn_b, wq_b, bq, nullptr, nullptr, q_b, MT_, D_, D_);
    gemm_wmma_kernel<0, false, false, true><<<gD, 256, 0, stream>>>(
        xn_b, wk_b, bk, nullptr, nullptr, k_b, MT_, D_, D_);
    gemm_wmma_kernel<0, false, false, true><<<gD, 256, 0, stream>>>(
        xn_b, wv_b, bv, nullptr, nullptr, v_b, MT_, D_, D_);

    // RoPE on q and k
    int rope_threads = MT_ * H_ * (HD_ / 2);
    rope_kernel<<<rope_threads / 256, 256, 0, stream>>>(q_b, freqs);
    rope_kernel<<<rope_threads / 256, 256, 0, stream>>>(k_b, freqs);

    // attention
    attn_kernel<<<dim3(S_ / 16, B_ * H_), 32, 0, stream>>>(q_b, k_b, v_b, pad, at_b);

    // O-projection + residual -> x2 (fp32)
    gemm_wmma_kernel<0, true, true, false><<<gD, 256, 0, stream>>>(
        at_b, wo_b, bo, x, x2_f, nullptr, MT_, D_, D_);

    // LN2
    ln_kernel<<<MT_, 256, 0, stream>>>(x2_f, ln2_g, ln2_b, h_b);

    // FFN1 (GELU)
    dim3 gF(FF_ / 64, MT_ / 128);
    gemm_wmma_kernel<1, false, false, true><<<gF, 256, 0, stream>>>(
        h_b, w1_b, b1, nullptr, nullptr, g_b, MT_, FF_, D_);

    // FFN2 + residual -> output (fp32)
    gemm_wmma_kernel<0, true, true, false><<<gD, 256, 0, stream>>>(
        g_b, w2_b, b2, x2_f, (float*)d_out, nullptr, MT_, D_, FF_);
}
